// AttentionAggregator_43585328120381
// MI455X (gfx1250) — matrix operand order repaired and checked by reading, not compiled, not run on hardware
//
#include <hip/hip_runtime.h>
#include <hip/hip_bf16.h>

// ---------------------------------------------------------------------------
// GAT-style attention aggregator for MI455X (gfx1250, wave32, WMMA).
// Main GEMMs run on v_wmma_f32_16x16x32_bf16 (f32 accumulate).
// Memory floor: ~330 MB of random feature gathers -> ~15us @ 23.3 TB/s;
// ~23 GFLOP of GEMM is absorbed by the matrix pipes, so we stay HBM-bound.
// Note: manual RNE bit-twiddle f32->bf16 is cheaper on this toolchain than
// both native __bf16 casts and __builtin_amdgcn_cvt_pk_bf16_f32 (probed).
// ---------------------------------------------------------------------------

typedef __attribute__((ext_vector_type(16))) __bf16 v16bf;
typedef __attribute__((ext_vector_type(8)))  float  v8f;
typedef __attribute__((ext_vector_type(4)))  unsigned int uintx4;
typedef __attribute__((ext_vector_type(2)))  unsigned int uintx2;

#define DDIM 128
#define KNB  32      // neighbors per node
#define NB   4       // target nodes per workgroup
#define SA   136     // ushorts per LDS A-row (272 B -> conflict-free A-operand loads)
#define ROWS_A (NB*KNB + 16)   // 128 neigh rows + one 16-row node tile
#define LEAKY 0.2f

union V16U { uintx4 q[2]; v16bf v; };

// f32 pair -> packed bf16 dword (round-to-nearest-even, manual: ~3 VALU/value)
__device__ __forceinline__ unsigned int pack2bf(float x, float y) {
  unsigned int a = __float_as_uint(x);
  unsigned int b = __float_as_uint(y);
  a += 0x7FFFu + ((a >> 16) & 1u);
  b += 0x7FFFu + ((b >> 16) & 1u);
  return (a >> 16) | (b & 0xFFFF0000u);
}
__device__ __forceinline__ unsigned short f2bf(float f) {
  unsigned int u = __float_as_uint(f);
  u += 0x7FFFu + ((u >> 16) & 1u);
  return (unsigned short)(u >> 16);
}
__device__ __forceinline__ float bf2f_lo(unsigned int u) {
  return __uint_as_float(u << 16);
}
__device__ __forceinline__ float bf2f_hi(unsigned int u) {
  return __uint_as_float(u & 0xFFFF0000u);
}
__device__ __forceinline__ v8f zero8() {
  v8f z = {0.f,0.f,0.f,0.f,0.f,0.f,0.f,0.f};
  return z;
}

// A operand (16x32 bf16 tile) from LDS. 16-bit A layout (ISA 7.12.2):
// lane l<16: row l,    K chunks at bytes [0..15] and [32..47]   (per 64B K-step)
// lane l>=16: row l-16, K chunks at bytes [16..31] and [48..63]
__device__ __forceinline__ v16bf loadA(const unsigned short* base, int row0,
                                       int kk, int r16, int hi) {
  int o = (row0 + r16) * SA + kk * 32 + hi * 8;   // ushort offset (mult of 8)
  V16U u;
  u.q[0] = *reinterpret_cast<const uintx4*>(base + o);
  u.q[1] = *reinterpret_cast<const uintx4*>(base + o + 16);
  return u.v;
}

// B operand (32x16 bf16 tile) from global W^T [128 out][128 in] bf16.
// B layout: lanes 0-15 = col n with K=0..7,16..23; lanes 16-31 = col n with K=8..15,24..31.
__device__ __forceinline__ v16bf loadB(const unsigned short* base, int col0,
                                       int kk, int r16, int hi) {
  int o = (col0 + r16) * DDIM + kk * 32 + hi * 8;
  V16U u;
  u.q[0] = *reinterpret_cast<const uintx4*>(base + o);
  u.q[1] = *reinterpret_cast<const uintx4*>(base + o + 16);
  return u.v;
}

// --------------------------- weight prep kernel ----------------------------
// Converts kernel, kernel1, neigh_weights (f32, [in][out]) to bf16 transposed
// [out][in] in workspace so B operands are contiguous 16B chunks.
__global__ __launch_bounds__(256)
void prep_weights(const float* __restrict__ k0, const float* __restrict__ k1,
                  const float* __restrict__ nw,
                  unsigned short* __restrict__ o0,
                  unsigned short* __restrict__ o1,
                  unsigned short* __restrict__ o2) {
  int idx = blockIdx.x * 256 + threadIdx.x;      // 3 * 16384
  int which = idx >> 14;
  int j = idx & 16383;
  int d = j >> 7;     // input dim
  int e = j & 127;    // output dim
  const float* src = (which == 0) ? k0 : ((which == 1) ? k1 : nw);
  unsigned short* dst = (which == 0) ? o0 : ((which == 1) ? o1 : o2);
  dst[e * DDIM + d] = f2bf(src[d * DDIM + e]);
}

// ------------------------------ main kernel --------------------------------
__global__ __launch_bounds__(256)
void gat_wmma_kernel(const float* __restrict__ features,
                     const int* __restrict__ node,
                     const int* __restrict__ neigh,
                     const unsigned short* __restrict__ WkT,   // kernel^T  bf16
                     const unsigned short* __restrict__ Wk1T,  // kernel1^T bf16
                     const unsigned short* __restrict__ WoT,   // neigh_w^T bf16
                     const float* __restrict__ aw,             // [256] f32
                     float* __restrict__ out, int Btot) {
  __shared__ __attribute__((aligned(16))) unsigned short ldsA[ROWS_A * SA];   // 39168 B
  __shared__ __attribute__((aligned(16))) unsigned short ldsNH[NB * KNB * SA];// 34816 B
  __shared__ __attribute__((aligned(16))) unsigned short ldsAF[16 * SA];      //  4352 B
  __shared__ float ldsNodeH[NB * DDIM];   // node_h rows
  __shared__ float ldsAW[2 * DDIM];       // attention weights
  __shared__ float ldsNS[NB];             // node part of score
  __shared__ float ldsScore[NB * KNB];
  __shared__ float ldsFS[NB * KNB];

  const int tid  = threadIdx.x;
  const int wave = tid >> 5;
  const int lane = tid & 31;
  const int hi   = lane >> 4;
  const int r16  = lane & 15;
  const int b0   = blockIdx.x * NB;

  ldsAW[tid] = aw[tid];                   // 256 threads -> 256 floats

  // ---- gather node + neighbor features -> bf16 LDS (coalesced b128 loads)
  for (int row = wave; row < NB * KNB + NB; row += 8) {
    int gb = (row < NB * KNB) ? (b0 + (row >> 5)) : (b0 + row - NB * KNB);
    uintx2 p = {0u, 0u};
    if (gb < Btot) {
      int src = (row < NB * KNB) ? neigh[b0 * KNB + row]
                                 : node[b0 + row - NB * KNB];
      const float4* sp =
          reinterpret_cast<const float4*>(features + (size_t)src * DDIM) + lane;
      float4 f = *sp;
      p.x = pack2bf(f.x, f.y);
      p.y = pack2bf(f.z, f.w);
    }
    *reinterpret_cast<uintx2*>(&ldsA[row * SA + lane * 4]) = p;  // one b64 store
  }
  __syncthreads();

  // ---- node_h = node_feat @ kernel : one 16-row tile (rows 128..143)
  {
    v8f acc = zero8();
#pragma unroll
    for (int kk = 0; kk < 4; ++kk) {
      v16bf a = loadA(ldsA, NB * KNB, kk, r16, hi);
      v16bf b = loadB(WkT, wave * 16, kk, r16, hi);
      acc = __builtin_amdgcn_wmma_f32_16x16x32_bf16(false, a, false, b,
                                                    (short)0, acc, false, false);
    }
    if (hi == 0) {                         // rows 0..NB-1 live in vgprs 0..3
#pragma unroll
      for (int v = 0; v < NB; ++v)
        ldsNodeH[v * DDIM + wave * 16 + r16] = acc[v];
    }
  }

  // ---- neigh_h = neigh_feat @ kernel1 : 8 M-tiles x 4 K-steps per wave
  {
    v8f accT[8];
#pragma unroll
    for (int m = 0; m < 8; ++m) accT[m] = zero8();
#pragma unroll
    for (int kk = 0; kk < 4; ++kk) {
      v16bf b = loadB(Wk1T, wave * 16, kk, r16, hi);   // cached across M tiles
#pragma unroll
      for (int m = 0; m < 8; ++m) {
        v16bf a = loadA(ldsA, m * 16, kk, r16, hi);
        accT[m] = __builtin_amdgcn_wmma_f32_16x16x32_bf16(
            false, a, false, b, (short)0, accT[m], false, false);
      }
    }
    // store neigh_h as bf16 (C layout: lane -> col r16, row m*16 + hi*8 + v)
#pragma unroll
    for (int m = 0; m < 8; ++m)
#pragma unroll
      for (int v = 0; v < 8; ++v)
        ldsNH[(m * 16 + hi * 8 + v) * SA + wave * 16 + r16] = f2bf(accT[m][v]);
  }
  __syncthreads();

  // ---- attention scores: dot(neigh_h, aw[0:128]) + dot(node_h, aw[128:256])
  float dotn = 0.f;
  if (tid < NB * KNB) {
    const uintx4* rp = reinterpret_cast<const uintx4*>(&ldsNH[tid * SA]);
#pragma unroll
    for (int c8 = 0; c8 < DDIM / 8; ++c8) {            // 16 x ds_load_b128
      uintx4 q = rp[c8];
      const float* a = &ldsAW[c8 * 8];
      dotn += bf2f_lo(q.x) * a[0] + bf2f_hi(q.x) * a[1];
      dotn += bf2f_lo(q.y) * a[2] + bf2f_hi(q.y) * a[3];
      dotn += bf2f_lo(q.z) * a[4] + bf2f_hi(q.z) * a[5];
      dotn += bf2f_lo(q.w) * a[6] + bf2f_hi(q.w) * a[7];
    }
  } else if (tid < NB * KNB + NB) {
    int b = tid - NB * KNB;
    float s = 0.f;
#pragma unroll 8
    for (int e = 0; e < DDIM; ++e)
      s += ldsNodeH[b * DDIM + e] * ldsAW[DDIM + e];
    ldsNS[b] = s;
  }
  __syncthreads();
  if (tid < NB * KNB) {
    float s = dotn + ldsNS[tid >> 5];
    ldsScore[tid] = (s > 0.f) ? s : LEAKY * s;     // leaky_relu(0.2)
  }
  __syncthreads();

  // ---- softmax over K=32 neighbors: one wave per target node
  if (wave < NB) {
    float s = ldsScore[wave * 32 + lane];
    float mx = s;
#pragma unroll
    for (int m = 16; m >= 1; m >>= 1) mx = fmaxf(mx, __shfl_xor(mx, m, 32));
    float e = __expf(s - mx);
    float sm = e;
#pragma unroll
    for (int m = 16; m >= 1; m >>= 1) sm += __shfl_xor(sm, m, 32);
    ldsFS[wave * 32 + lane] = e / sm;
  }
  __syncthreads();

  // ---- weighted aggregation: attn_feat[b][e] = sum_k nh[b,k,e] * fs[b,k]
  // dword granularity: thread handles (b, pair-of-columns)
  {
    int b  = tid >> 6;            // 0..3
    int e2 = tid & 63;            // dword column pair
    float af0 = 0.f, af1 = 0.f;
    const unsigned int* col =
        reinterpret_cast<const unsigned int*>(&ldsNH[b * KNB * SA + e2 * 2]);
#pragma unroll
    for (int k = 0; k < KNB; ++k) {
      unsigned int q = col[k * (SA / 2)];
      float w = ldsFS[b * KNB + k];
      af0 += bf2f_lo(q) * w;
      af1 += bf2f_hi(q) * w;
    }
    *reinterpret_cast<unsigned int*>(&ldsAF[b * SA + e2 * 2]) =
        pack2bf(af0, af1);
  }
  for (int i = tid; i < (16 - NB) * DDIM / 2; i += 256) {  // zero pad rows 4..15
    int rr = NB + i / 64, e2 = i & 63;
    *reinterpret_cast<unsigned int*>(&ldsAF[rr * SA + e2 * 2]) = 0u;
  }
  __syncthreads();

  // ---- output projection: attn_feat @ neigh_weights (one M-tile per wave-N)
  {
    v8f acc = zero8();
#pragma unroll
    for (int kk = 0; kk < 4; ++kk) {
      v16bf a = loadA(ldsAF, 0, kk, r16, hi);
      v16bf b = loadB(WoT, wave * 16, kk, r16, hi);
      acc = __builtin_amdgcn_wmma_f32_16x16x32_bf16(false, a, false, b,
                                                    (short)0, acc, false, false);
    }
    if (hi == 0) {
#pragma unroll
      for (int v = 0; v < NB; ++v)
        if (b0 + v < Btot)
          out[(size_t)(b0 + v) * DDIM + wave * 16 + r16] = acc[v];
    }
  }
}

// ------------------------------- launcher ----------------------------------
extern "C" void kernel_launch(void* const* d_in, const int* in_sizes, int n_in,
                              void* d_out, int out_size, void* d_ws, size_t ws_size,
                              hipStream_t stream) {
  const float* features = (const float*)d_in[0];
  const int*   node     = (const int*)d_in[1];
  const int*   neigh    = (const int*)d_in[2];
  const float* k0       = (const float*)d_in[3];   // kernel  [1,128,128]
  const float* k1       = (const float*)d_in[4];   // kernel1 [1,128,128]
  const float* aw       = (const float*)d_in[5];   // attention_weights [1,256]
  const float* nw       = (const float*)d_in[6];   // neigh_weights [128,128]

  unsigned short* ws = (unsigned short*)d_ws;      // 3 * 32 KB bf16 weights
  unsigned short* wkT  = ws;
  unsigned short* wk1T = ws + 16384;
  unsigned short* woT  = ws + 32768;

  prep_weights<<<192, 256, 0, stream>>>(k0, k1, nw, wkT, wk1T, woT);

  int Btot = in_sizes[1];                          // node is [B,1] -> B ints
  int grid = (Btot + NB - 1) / NB;                 // 5000 for B=20000
  gat_wmma_kernel<<<grid, 256, 0, stream>>>(features, node, neigh,
                                            wkT, wk1T, woT, aw,
                                            (float*)d_out, Btot);
}
